// CrossAttentionLayer_30039001268913
// MI455X (gfx1250) — matrix operand order repaired and checked by reading, not compiled, not run on hardware
//
#include <hip/hip_runtime.h>
#include <hip/hip_bf16.h>

#define BQ   4
#define SEQ  2048
#define DIMD 1024

typedef __attribute__((ext_vector_type(16))) __bf16       v16bf;
typedef __attribute__((ext_vector_type(8)))  float        v8f;
typedef __attribute__((ext_vector_type(4)))  unsigned int v4u;

// ---------- WMMA fragment loaders (bf16, 16x16x32) ----------
// A (16x32, MxK): lane L holds row M=L%16. Lanes 0-15: K {0..7,16..23};
// lanes 16-31: K {8..15,24..31}. Two 16B loads.
__device__ __forceinline__ v16bf load_a_frag(const __bf16* base, int ld,
                                             int row, int k0, int lane) {
  const int koff = (lane & 16) ? 8 : 0;
  const __bf16* p = base + (size_t)row * ld + k0 + koff;
  union { v4u q[2]; v16bf f; } u;
  u.q[0] = *reinterpret_cast<const v4u*>(p);
  u.q[1] = *reinterpret_cast<const v4u*>(p + 16);
  return u.f;
}
// B (32x16, KxN): lane L holds column N=L%16 (a row of W when W is [N][K]).
// Lanes 0-15: K {0..7,16..23}; lanes 16-31: K {8..15,24..31}.
__device__ __forceinline__ v16bf load_b_frag(const __bf16* base, int ld,
                                             int row, int k0, int lane) {
  const int koff = (lane & 16) ? 8 : 0;
  const __bf16* p = base + (size_t)row * ld + k0 + koff;
  union { v4u q[2]; v16bf f; } u;
  u.q[0] = *reinterpret_cast<const v4u*>(p);
  u.q[1] = *reinterpret_cast<const v4u*>(p + 16);
  return u.f;
}

// One k-slice (k..k+31) of a 32M x 64N wave tile: 2 A frags + 4 B frags.
struct Frags { v16bf a0, a1, b0, b1, b2, b3; };

__device__ __forceinline__ void load_tile(Frags& f,
                                          const __bf16* A, const __bf16* W,
                                          int lda, int ldw,
                                          int arow, int brow, int k0, int lane)
{
  f.a0 = load_a_frag(A, lda, arow,      k0, lane);
  f.a1 = load_a_frag(A, lda, arow + 16, k0, lane);
  f.b0 = load_b_frag(W, ldw, brow,      k0, lane);
  f.b1 = load_b_frag(W, ldw, brow + 16, k0, lane);
  f.b2 = load_b_frag(W, ldw, brow + 32, k0, lane);
  f.b3 = load_b_frag(W, ldw, brow + 48, k0, lane);
}

__device__ __forceinline__ void mma_tile(const Frags& f, v8f acc[2][4])
{
#define WMMA_(AF, J) \
  acc[(AF)][(J)] = __builtin_amdgcn_wmma_f32_16x16x32_bf16( \
      false, (AF) ? f.a1 : f.a0, false, \
      (J) == 0 ? f.b0 : (J) == 1 ? f.b1 : (J) == 2 ? f.b2 : f.b3, \
      (short)0, acc[(AF)][(J)], false, false)
  WMMA_(0, 0); WMMA_(0, 1); WMMA_(0, 2); WMMA_(0, 3);
  WMMA_(1, 0); WMMA_(1, 1); WMMA_(1, 2); WMMA_(1, 3);
#undef WMMA_
}

// ---------- Generic bf16 WMMA GEMM: C[m][n] = scale * sum_k A[m][k]*W[n][k] + bias[n]
// Block = 128 threads = 4 waves (2M x 2N). Wave tile = 32M x 64N.
// Block tile = 64M x 128N. grid = (N/128, M/64, nbatch). K % 64 == 0, K >= 64.
__global__ __launch_bounds__(128)
void wmma_gemm_bf16(const __hip_bfloat16* __restrict__ A_,
                    const __hip_bfloat16* __restrict__ W_,
                    const float* __restrict__ bias,
                    float* __restrict__ outF,
                    __hip_bfloat16* __restrict__ outB,
                    long long strideA, long long strideW, long long strideO,
                    int lda, int ldw, int ldo, int K, int transOut, float scale)
{
  const __bf16* A = reinterpret_cast<const __bf16*>(A_) + (long long)blockIdx.z * strideA;
  const __bf16* W = reinterpret_cast<const __bf16*>(W_) + (long long)blockIdx.z * strideW;
  const int lane = threadIdx.x & 31;
  const int wid  = threadIdx.x >> 5;
  const int m0   = blockIdx.y * 64 + (wid >> 1) * 32;
  const int n0   = blockIdx.x * 128 + (wid & 1) * 64;
  const int arow = m0 + (lane & 15);
  const int brow = n0 + (lane & 15);

  v8f acc[2][4] = {};
  Frags f0, f1;

  // Software-pipelined k loop: two 32-deep slices in flight.
  load_tile(f0, A, W, lda, ldw, arow, brow, 0, lane);
  for (int k0 = 0; k0 < K; k0 += 64) {
    load_tile(f1, A, W, lda, ldw, arow, brow, k0 + 32, lane);
    __builtin_prefetch(A + (size_t)arow * lda + k0 + 256, 0, 1);  // global_prefetch_b8
    mma_tile(f0, acc);
    const int kn = (k0 + 64 < K) ? (k0 + 64) : k0;  // clamped (dead) load on last iter
    load_tile(f0, A, W, lda, ldw, arow, brow, kn, lane);
    mma_tile(f1, acc);
  }

  // C/D layout: lane owns column n = n0+j*16+(lane&15); VGPR v -> row +v (+8 hi lanes)
  const int mhalf = (lane >> 4) << 3;
#pragma unroll
  for (int i = 0; i < 2; ++i) {
    const int mb = m0 + i * 16 + mhalf;
#pragma unroll
    for (int j = 0; j < 4; ++j) {
      const int n = n0 + j * 16 + (lane & 15);
      const float bv = bias ? bias[n] : 0.0f;
#pragma unroll
      for (int v = 0; v < 8; ++v) {
        const int m = mb + v;
        const float val = acc[i][j][v] * scale + bv;
        const long long idx = (long long)blockIdx.z * strideO +
            (transOut ? ((long long)n * ldo + m) : ((long long)m * ldo + n));
        if (outB) outB[idx] = __float2bfloat16(val);
        else      outF[idx] = val;
      }
    }
  }
}

// ---------- Row softmax: scores (fp32) -> attn (bf16) + attnT (bf16)
// One block per (q,b) row; async-copies the 8KB row to LDS via the CDNA5
// ASYNCcnt path, then reduces with wave32 shuffles + LDS.
__global__ __launch_bounds__(256)
void softmax_rows(const float* __restrict__ scores,
                  __hip_bfloat16* __restrict__ attn,
                  __hip_bfloat16* __restrict__ attnT,
                  float scale)
{
  __shared__ __align__(16) float row[SEQ];
  __shared__ float redmax[8];
  __shared__ float redsum[8];
  const int q = blockIdx.x, b = blockIdx.y;
  const int tid = threadIdx.x;
  const int lane = tid & 31, wave = tid >> 5;
  const float* src = scores + ((size_t)b * SEQ + q) * SEQ;

  // 32 bytes/thread: two global_load_async_to_lds_b128 (ASYNCcnt-tracked)
  {
    unsigned lds = (unsigned)(unsigned long long)(void*)&row[tid * 8];
    unsigned long long ga = (unsigned long long)(src + (size_t)tid * 8);
    asm volatile("global_load_async_to_lds_b128 %0, %1, off"
                 :: "v"(lds), "v"(ga) : "memory");
    asm volatile("global_load_async_to_lds_b128 %0, %1, off offset:16"
                 :: "v"(lds), "v"(ga) : "memory");
    asm volatile("s_wait_asynccnt 0" ::: "memory");
  }
  __syncthreads();

  float e[8];
  float mx = -3.0e38f;
#pragma unroll
  for (int i = 0; i < 8; ++i) {
    const float v = row[tid + i * 256] * scale;
    e[i] = v;
    mx = fmaxf(mx, v);
  }
#pragma unroll
  for (int off = 16; off >= 1; off >>= 1)
    mx = fmaxf(mx, __shfl_xor(mx, off, 32));
  if (lane == 0) redmax[wave] = mx;
  __syncthreads();
  float rmax = redmax[0];
#pragma unroll
  for (int w = 1; w < 8; ++w) rmax = fmaxf(rmax, redmax[w]);

  float s = 0.0f;
#pragma unroll
  for (int i = 0; i < 8; ++i) { e[i] = __expf(e[i] - rmax); s += e[i]; }
#pragma unroll
  for (int off = 16; off >= 1; off >>= 1) s += __shfl_xor(s, off, 32);
  if (lane == 0) redsum[wave] = s;
  __syncthreads();
  float rsum = 0.0f;
#pragma unroll
  for (int w = 0; w < 8; ++w) rsum += redsum[w];
  const float inv = 1.0f / rsum;

#pragma unroll
  for (int i = 0; i < 8; ++i) {
    const int col = tid + i * 256;
    const __hip_bfloat16 p = __float2bfloat16(e[i] * inv);
    attn [((size_t)b * SEQ + q)   * SEQ + col] = p;
    attnT[((size_t)b * SEQ + col) * SEQ + q]   = p;
  }
}

// ---------- Input split/convert: fp32 inputs -> vision bf16, text bf16, visionT bf16
__global__ __launch_bounds__(256)
void convert_inputs(const float* __restrict__ in,
                    __hip_bfloat16* __restrict__ vis,
                    __hip_bfloat16* __restrict__ txt,
                    __hip_bfloat16* __restrict__ visT)
{
  const size_t idx = (size_t)blockIdx.x * 256 + threadIdx.x;  // over B*SEQ*DIMD
  const size_t d  = idx % DIMD;
  const size_t rs = idx / DIMD;               // b*SEQ + s
  const size_t b  = rs / SEQ, s2 = rs % SEQ;
  const float v = in[rs * (2 * DIMD) + d];
  const float t = in[rs * (2 * DIMD) + DIMD + d];
  vis[idx] = __float2bfloat16(v);
  txt[idx] = __float2bfloat16(t);
  visT[d * (size_t)(BQ * SEQ) + b * SEQ + s2] = __float2bfloat16(v);  // [D][B*S]
}

__global__ __launch_bounds__(256)
void convert_weight(const float* __restrict__ w, __hip_bfloat16* __restrict__ o)
{
  const size_t idx = (size_t)blockIdx.x * 256 + threadIdx.x;
  o[idx] = __float2bfloat16(w[idx]);
}

extern "C" void kernel_launch(void* const* d_in, const int* in_sizes, int n_in,
                              void* d_out, int out_size, void* d_ws, size_t ws_size,
                              hipStream_t stream)
{
  const float* inputs = (const float*)d_in[0];
  const float* Wq = (const float*)d_in[1];
  const float* bq = (const float*)d_in[2];
  const float* Wk = (const float*)d_in[3];
  const float* bk = (const float*)d_in[4];
  const float* Wv = (const float*)d_in[5];
  const float* bv = (const float*)d_in[6];
  float* out = (float*)d_out;

  const int B = BQ, S = SEQ, D = DIMD;
  char* ws = (char*)d_ws;
  size_t off = 0;
  auto take = [&](size_t bytes) -> void* {
    void* p = ws + off;
    off = (off + bytes + 255) & ~(size_t)255;
    return p;
  };
  __hip_bfloat16* visBF = (__hip_bfloat16*)take((size_t)B * S * D * 2);
  __hip_bfloat16* txtBF = (__hip_bfloat16*)take((size_t)B * S * D * 2);
  __hip_bfloat16* visT  = (__hip_bfloat16*)take((size_t)B * S * D * 2);  // [D][B*S]
  __hip_bfloat16* wqBF  = (__hip_bfloat16*)take((size_t)D * D * 2);
  __hip_bfloat16* wkBF  = (__hip_bfloat16*)take((size_t)D * D * 2);
  __hip_bfloat16* wvBF  = (__hip_bfloat16*)take((size_t)D * D * 2);
  __hip_bfloat16* Qbf   = (__hip_bfloat16*)take((size_t)B * S * D * 2);  // [B*S][D]
  __hip_bfloat16* Kbf   = (__hip_bfloat16*)take((size_t)B * S * D * 2);  // [B*S][D]
  __hip_bfloat16* Vt    = (__hip_bfloat16*)take((size_t)B * S * D * 2);  // [D][B*S]
  __hip_bfloat16* attn  = (__hip_bfloat16*)take((size_t)B * S * S * 2);  // [B][q][k]
  __hip_bfloat16* attnT = (__hip_bfloat16*)take((size_t)B * S * S * 2);  // [B][k][q]
  float*          scr   = (float*)take((size_t)B * S * S * 4);           // [B][q][k]

  const dim3 blk256(256), blk128(128);

  convert_inputs<<<dim3((unsigned)(((size_t)B * S * D) / 256)), blk256, 0, stream>>>(
      inputs, visBF, txtBF, visT);
  convert_weight<<<dim3(D * D / 256), blk256, 0, stream>>>(Wq, wqBF);
  convert_weight<<<dim3(D * D / 256), blk256, 0, stream>>>(Wk, wkBF);
  convert_weight<<<dim3(D * D / 256), blk256, 0, stream>>>(Wv, wvBF);

  // Q = vision @ Wq^T + bq   (M=B*S, N=D, K=D)
  wmma_gemm_bf16<<<dim3(D / 128, (B * S) / 64, 1), blk128, 0, stream>>>(
      visBF, wqBF, bq, nullptr, Qbf, 0, 0, 0, D, D, D, D, 0, 1.0f);
  // K = text @ Wk^T + bk
  wmma_gemm_bf16<<<dim3(D / 128, (B * S) / 64, 1), blk128, 0, stream>>>(
      txtBF, wkBF, bk, nullptr, Kbf, 0, 0, 0, D, D, D, D, 0, 1.0f);
  // V^T = (text @ Wv^T + bv)^T stored as [D][B*S]
  wmma_gemm_bf16<<<dim3(D / 128, (B * S) / 64, 1), blk128, 0, stream>>>(
      txtBF, wvBF, bv, nullptr, Vt, 0, 0, 0, D, D, B * S, D, 1, 1.0f);

  // scores[b] = Q[b] @ K[b]^T  (M=S, N=S, K=D), scaled later in softmax
  wmma_gemm_bf16<<<dim3(S / 128, S / 64, B), blk128, 0, stream>>>(
      Qbf, Kbf, nullptr, scr, nullptr,
      (long long)S * D, (long long)S * D, (long long)S * S, D, D, S, D, 0, 1.0f);

  // softmax(scores / sqrt(D)) -> attn + attn^T (bf16)
  softmax_rows<<<dim3(S, B), blk256, 0, stream>>>(scr, attn, attnT, 0.03125f);

  // cross_attended_vision[b] = attn[b] @ V[b]   (W = Vt[:, b*S:(b+1)*S], ldw = B*S)
  wmma_gemm_bf16<<<dim3(D / 128, S / 64, B), blk128, 0, stream>>>(
      attn, Vt, nullptr, out, nullptr,
      (long long)S * S, (long long)S, (long long)S * D, S, B * S, D, S, 0, 1.0f);

  // cross_attended_text[b] = attn[b]^T @ vision[b]  (A = attnT, W = visT columns)
  wmma_gemm_bf16<<<dim3(D / 128, S / 64, B), blk128, 0, stream>>>(
      attnT, visT, nullptr, out + (size_t)B * S * D, nullptr,
      (long long)S * S, (long long)S, (long long)S * D, S, B * S, D, S, 0, 1.0f);
}
